// SetwiseDistance_origin_75222057222593
// MI455X (gfx1250) — matrix operand order repaired and checked by reading, not compiled, not run on hardware
//
#include <hip/hip_runtime.h>

typedef __attribute__((ext_vector_type(16))) _Float16 v16h;
typedef __attribute__((ext_vector_type(8)))  _Float16 v8h;
typedef __attribute__((ext_vector_type(8)))  float    v8f;
typedef __attribute__((ext_vector_type(4)))  int      v4i;

#define D_DIM 256
#define TEMP_S 16.0f

#define HAS_ASYNC_LDS __has_builtin(__builtin_amdgcn_global_load_async_to_lds_b128)

// ---------------------------------------------------------------------------
// f32 -> f16 pack (both embedding matrices), vectorized 8 elems / thread
// ---------------------------------------------------------------------------
__global__ void cvt_f32_to_f16_kernel(const float* __restrict__ src,
                                      _Float16* __restrict__ dst, int n) {
  int i = (blockIdx.x * blockDim.x + threadIdx.x) * 8;
  if (i < n) {
    const float4* s4 = (const float4*)(src + i);
    float4 f0 = s4[0];
    float4 f1 = s4[1];
    v8h h = { (_Float16)f0.x, (_Float16)f0.y, (_Float16)f0.z, (_Float16)f0.w,
              (_Float16)f1.x, (_Float16)f1.y, (_Float16)f1.z, (_Float16)f1.w };
    *(v8h*)(dst + i) = h;
  }
}

// 16-byte async global->LDS copy (CDNA5 GLOBAL_LOAD_ASYNC_TO_LDS_B128,
// tracked by ASYNCcnt); synchronous fallback if the builtin is absent.
__device__ __forceinline__ void async_copy_b128(const _Float16* g, _Float16* l) {
#if HAS_ASYNC_LDS
  void* gp = (void*)g;  // drop const, keep generic
  void* lp = (void*)l;
  __builtin_amdgcn_global_load_async_to_lds_b128(
      (__attribute__((address_space(1))) v4i*)gp,
      (__attribute__((address_space(3))) v4i*)lp, 0, 0);
#else
  *(v8h*)l = *(const v8h*)g;
#endif
}

__device__ __forceinline__ void wait_async_copies() {
#if HAS_ASYNC_LDS
#if __has_builtin(__builtin_amdgcn_s_wait_asynccnt)
  __builtin_amdgcn_s_wait_asynccnt(0);
#else
  asm volatile("s_wait_asynccnt 0x0" ::: "memory");
#endif
#endif
}

// in-lane log-sum-exp over 4 values (inputs already scaled by TEMP)
__device__ __forceinline__ float lse4(float a, float b, float c, float d) {
  float m = fmaxf(fmaxf(a, b), fmaxf(c, d));
  float s = __expf(a - m) + __expf(b - m) + __expf(c - m) + __expf(d - m);
  return m + __logf(s);
}

// ---------------------------------------------------------------------------
// Fused WMMA GEMM + setwise LSE epilogue.
// Block = 8 waves -> 128x64 dist macro-tile. The shared 64-row txt strip
// (contiguous 32 KB) is staged once per block into LDS with async copies;
// each wave preloads its 16x256 img panel into registers, then the K-loop
// is pure ds_load_b128 + v_wmma.
// ---------------------------------------------------------------------------
__global__ __launch_bounds__(256) void setwise_wmma_kernel(
    const _Float16* __restrict__ A,  // img f16 [NI][256] row-major
    const _Float16* __restrict__ B,  // txt f16 [NT][256] row-major
    float* __restrict__ out, int NI, int NT) {
  alignas(16) __shared__ _Float16 ldsB[64 * D_DIM];  // 32 KB

  const int lane = threadIdx.x;              // 0..31
  const int wave = threadIdx.y;              // 0..7
  const int mTile  = blockIdx.y * 8 + wave;  // 16-row tile of dist
  const int nStrip = blockIdx.x;             // 64-col strip of dist
  const int R0 = mTile * 16;
  const int C0 = nStrip * 64;

  // ---- stage B strip (txt rows C0..C0+63, flat 32 KB) into LDS ----
  {
    const int tid = wave * 32 + lane;        // 0..255
    const _Float16* gsrc = B + (size_t)C0 * D_DIM;
#pragma unroll
    for (int i = 0; i < 8; ++i) {
      const int h8 = (i * 256 + tid) * 8;    // 16-byte chunk -> half offset
      async_copy_b128(gsrc + h8, ldsB + h8);
    }
  }

  const int l15 = lane & 15;
  const int hi  = lane >> 4;  // lane half

  // ---- preload this wave's full A panel (16 rows x 256 K) into registers ----
  // A fragment (16x32 f16): low lanes K 0-7 (v0-3) + 16-23 (v4-7);
  //                         high lanes K 8-15 + 24-31. (ISA 7.12.2)
  const _Float16* arow = A + (size_t)(R0 + l15) * D_DIM + hi * 8;
  v16h afrag[8];
#pragma unroll
  for (int kc8 = 0; kc8 < 8; ++kc8) {
    v8h lo = *(const v8h*)(arow + kc8 * 32);
    v8h hh = *(const v8h*)(arow + kc8 * 32 + 16);
#pragma unroll
    for (int j = 0; j < 8; ++j) { afrag[kc8][j] = lo[j]; afrag[kc8][8 + j] = hh[j]; }
  }

  wait_async_copies();
  __syncthreads();

  // B fragment (32x16 f16): lane = column; low lanes K 0-7 + 16-23, high lanes K 8-15 + 24-31.
  const _Float16* brow0 = ldsB + (size_t)l15 * D_DIM + hi * 8;

  v8f acc[4] = {v8f{}, v8f{}, v8f{}, v8f{}};
#pragma unroll
  for (int kc8 = 0; kc8 < 8; ++kc8) {
    const int kc = kc8 * 32;
#pragma unroll
    for (int nt = 0; nt < 4; ++nt) {
      const _Float16* brow = brow0 + (size_t)(nt * 16) * D_DIM;
      v16h b;
      {
        v8h lo = *(const v8h*)(brow + kc);
        v8h hh = *(const v8h*)(brow + kc + 16);
#pragma unroll
        for (int j = 0; j < 8; ++j) { b[j] = lo[j]; b[8 + j] = hh[j]; }
      }
      acc[nt] = __builtin_amdgcn_wmma_f32_16x16x32_f16(
          false, afrag[kc8], false, b, (short)0, acc[nt], false, false);
    }
  }

  // ---- fused setwise-LSE epilogue ----
  // C/D layout: lane (0-15 | 16-31) holds column N = lane&15,
  // VGPR v holds row M = v (low lanes) or v+8 (high lanes).
  const float inv = 1.0f / (2.0f * 4.0f * TEMP_S);  // /(IMG_SET*s)/DENOM
  const int NTo = NT / 4;
  const int OIbase = R0 / 4 + hi * 2;   // output row group base for this lane
  const int sub = lane & 3;

#pragma unroll
  for (int nt = 0; nt < 4; ++nt) {
    v8f c = acc[nt];
    float x[8];
#pragma unroll
    for (int j = 0; j < 8; ++j) x[j] = TEMP_S * c[j];

    // left: LSE over 4 img rows (in-lane, across accumulator VGPRs),
    // then sum over the 4 txt cols of the group (cross-lane sum, xor 1|2).
    float l_lo = lse4(x[0], x[1], x[2], x[3]);
    float l_hi = lse4(x[4], x[5], x[6], x[7]);
    l_lo += __shfl_xor(l_lo, 1, 32);
    l_lo += __shfl_xor(l_lo, 2, 32);
    l_hi += __shfl_xor(l_hi, 1, 32);
    l_hi += __shfl_xor(l_hi, 2, 32);

    // right: per row, LSE over the 4 txt cols (cross-lane), summed over rows.
    float r_lo = 0.0f, r_hi = 0.0f;
#pragma unroll
    for (int j = 0; j < 8; ++j) {
      float m = x[j];
      m = fmaxf(m, __shfl_xor(m, 1, 32));
      m = fmaxf(m, __shfl_xor(m, 2, 32));
      float e = __expf(x[j] - m);
      e += __shfl_xor(e, 1, 32);
      e += __shfl_xor(e, 2, 32);
      float r = m + __logf(e);
      if (j < 4) r_lo += r; else r_hi += r;
    }

    // s_r == s_l == 16, both set sizes == 4 -> out = (right + left) / 128
    float out_lo = (r_lo + l_lo) * inv;
    float out_hi = (r_hi + l_hi) * inv;

    const int OJ = (C0 + nt * 16) / 4 + (l15 >> 2);
    if (sub == 0) {
      out[(size_t)OIbase * NTo + OJ] = out_lo;
    } else if (sub == 1) {
      out[(size_t)(OIbase + 1) * NTo + OJ] = out_hi;
    }
  }
}

extern "C" void kernel_launch(void* const* d_in, const int* in_sizes, int n_in,
                              void* d_out, int out_size, void* d_ws, size_t ws_size,
                              hipStream_t stream) {
  const float* img = (const float*)d_in[0];
  const float* txt = (const float*)d_in[1];
  float* out = (float*)d_out;

  const int NI = in_sizes[0] / D_DIM;  // 8192
  const int NT = in_sizes[1] / D_DIM;  // 8192

  _Float16* wsA = (_Float16*)d_ws;
  _Float16* wsB = wsA + (size_t)NI * D_DIM;

  const int nA = NI * D_DIM;
  const int nB = NT * D_DIM;
  cvt_f32_to_f16_kernel<<<(nA / 8 + 255) / 256, 256, 0, stream>>>(img, wsA, nA);
  cvt_f32_to_f16_kernel<<<(nB / 8 + 255) / 256, 256, 0, stream>>>(txt, wsB, nB);

  dim3 block(32, 8);                       // 8 waves per block
  dim3 grid(NT / 64, NI / (16 * 8));       // 128 x 64 blocks
  setwise_wmma_kernel<<<grid, block, 0, stream>>>(wsA, wsB, out, NI, NT);
}
